// PairformerStack_64441689309766
// MI455X (gfx1250) — compile-verified, matrix operand order checked by reading, not run on hardware
//
#include <hip/hip_runtime.h>
#include <cstdint>

typedef long long i64;
typedef unsigned short u16;
typedef __attribute__((ext_vector_type(16))) __bf16 v16bf;
typedef __attribute__((ext_vector_type(8)))  __bf16 v8bf;
typedef __attribute__((ext_vector_type(8)))  float  v8f;

// ---------------- device helpers ----------------
__device__ inline u16 f2bf(float f) {
  union { float f; unsigned u; } v; v.f = f;
  unsigned r = v.u + 0x7FFFu + ((v.u >> 16) & 1u);
  if ((v.u & 0x7F800000u) == 0x7F800000u) r = v.u;   // inf/nan passthrough
  return (u16)(r >> 16);
}
__device__ inline float sigmoidf_(float x) { return 1.0f / (1.0f + __expf(-x)); }

#define SHUF16(lo, hi) __builtin_shufflevector((lo), (hi), 0, 1, 2, 3, 4, 5, 6, 7, 8, 9, 10, 11, 12, 13, 14, 15)

// ---------------- generic batched strided bf16 WMMA GEMM (vector path only) ----------------
// C[b2][b1][m][n] (fp32) = sum_k A * B.
// Requirements (guaranteed by all call sites): A and B are k-contiguous
// (element (m,k) at A + aoff + m*am + k ; element (k,n) at B + boff + n*bn + k),
// K % 32 == 0, all strides/offsets multiples of 8 elements (16B alignment).
__global__ __launch_bounds__(256)
void k_gemm_bf16(const u16* __restrict__ A, i64 am, i64 ab1, i64 ab2,
                 const u16* __restrict__ B, i64 bn, i64 bb1, i64 bb2,
                 float* __restrict__ C, i64 cm, i64 cn, i64 cb1, i64 cb2,
                 int M, int N, int K, int NB1)
{
  __shared__ __align__(16) u16 As[64][40];
  __shared__ __align__(16) u16 Bs[64][40];   // [n][k]
  int bz = blockIdx.z;
  int b1 = bz % NB1, b2 = bz / NB1;
  i64 aoff = (i64)b1 * ab1 + (i64)b2 * ab2;
  i64 boff = (i64)b1 * bb1 + (i64)b2 * bb2;
  i64 coff = (i64)b1 * cb1 + (i64)b2 * cb2;
  int m0 = blockIdx.x * 64, n0 = blockIdx.y * 64;
  int tid = threadIdx.x, lane = tid & 31, wave = tid >> 5;
  int wm = wave & 3, wn = wave >> 2;          // wave tile: rows [wm*16,+16), cols [wn*32,+32)
  int srow = tid >> 2, schk = (tid & 3) * 8;  // 64 rows x 4 chunks of 8 bf16
  v8f c0 = {}, c1 = {};
  for (int k0 = 0; k0 < K; k0 += 32) {
    int gm = m0 + srow;
    uint4 va = make_uint4(0u, 0u, 0u, 0u);
    if (gm < M) va = *(const uint4*)(A + aoff + (i64)gm * am + (k0 + schk));
    *(uint4*)&As[srow][schk] = va;
    int gn = n0 + srow;
    uint4 vb = make_uint4(0u, 0u, 0u, 0u);
    if (gn < N) vb = *(const uint4*)(B + boff + (i64)gn * bn + (k0 + schk));
    *(uint4*)&Bs[srow][schk] = vb;
    if (k0 + 32 < K && gm < M)
      __builtin_prefetch(A + aoff + (i64)gm * am + (k0 + 32 + schk), 0, 1);
    __syncthreads();
    int ml = lane & 15, hi = lane >> 4;
    const v8bf* ar  = (const v8bf*)&As[wm * 16 + ml][0];
    const v8bf* br0 = (const v8bf*)&Bs[wn * 32 + ml][0];
    const v8bf* br1 = (const v8bf*)&Bs[wn * 32 + 16 + ml][0];
    v16bf fa  = SHUF16(ar[hi],  ar[2 + hi]);
    v16bf fb0 = SHUF16(br0[hi], br0[2 + hi]);
    v16bf fb1 = SHUF16(br1[hi], br1[2 + hi]);
    c0 = __builtin_amdgcn_wmma_f32_16x16x32_bf16(false, fa, false, fb0, (short)0, c0, false, false);
    c1 = __builtin_amdgcn_wmma_f32_16x16x32_bf16(false, fa, false, fb1, (short)0, c1, false, false);
    __syncthreads();
  }
  int nl = lane & 15, hi = lane >> 4;
#pragma unroll
  for (int r = 0; r < 8; ++r) {
    int gm = m0 + wm * 16 + r + hi * 8;
    if (gm >= M) continue;
    int gn0 = n0 + wn * 32 + nl;
    if (gn0 < N) C[coff + (i64)gm * cm + (i64)gn0 * cn] = c0[r];
    int gn1 = gn0 + 16;
    if (gn1 < N) C[coff + (i64)gm * cm + (i64)gn1 * cn] = c1[r];
  }
}

// ---------------- fused dual-GEMM SwiGLU: H = silu(A@W1) * (A@W2) (bf16 out) ----------------
// Weights pre-transposed: W[n*K + k]. A contiguous in k.
__global__ __launch_bounds__(256)
void k_gemm_swiglu(const u16* __restrict__ A, i64 am,
                   const u16* __restrict__ W1, const u16* __restrict__ W2,
                   u16* __restrict__ H, i64 hm,
                   int M, int N, int K)
{
  __shared__ __align__(16) u16 As[64][40];
  __shared__ __align__(16) u16 W1s[64][40];
  __shared__ __align__(16) u16 W2s[64][40];
  int m0 = blockIdx.x * 64, n0 = blockIdx.y * 64;
  int tid = threadIdx.x, lane = tid & 31, wave = tid >> 5;
  int wm = wave & 3, wn = wave >> 2;
  int srow = tid >> 2, schk = (tid & 3) * 8;
  v8f c10 = {}, c11 = {}, c20 = {}, c21 = {};
  for (int k0 = 0; k0 < K; k0 += 32) {
    int gm = m0 + srow;
    uint4 va = make_uint4(0u, 0u, 0u, 0u);
    if (gm < M) va = *(const uint4*)(A + (i64)gm * am + (k0 + schk));
    *(uint4*)&As[srow][schk] = va;
    int gn = n0 + srow;
    uint4 v1 = make_uint4(0u, 0u, 0u, 0u), v2 = make_uint4(0u, 0u, 0u, 0u);
    if (gn < N) {
      v1 = *(const uint4*)(W1 + (i64)gn * K + (k0 + schk));
      v2 = *(const uint4*)(W2 + (i64)gn * K + (k0 + schk));
    }
    *(uint4*)&W1s[srow][schk] = v1;
    *(uint4*)&W2s[srow][schk] = v2;
    __syncthreads();
    int ml = lane & 15, hi = lane >> 4;
    const v8bf* ar  = (const v8bf*)&As[wm * 16 + ml][0];
    const v8bf* w10 = (const v8bf*)&W1s[wn * 32 + ml][0];
    const v8bf* w11 = (const v8bf*)&W1s[wn * 32 + 16 + ml][0];
    const v8bf* w20 = (const v8bf*)&W2s[wn * 32 + ml][0];
    const v8bf* w21 = (const v8bf*)&W2s[wn * 32 + 16 + ml][0];
    v16bf fa  = SHUF16(ar[hi],  ar[2 + hi]);
    v16bf f10 = SHUF16(w10[hi], w10[2 + hi]);
    v16bf f11 = SHUF16(w11[hi], w11[2 + hi]);
    v16bf f20 = SHUF16(w20[hi], w20[2 + hi]);
    v16bf f21 = SHUF16(w21[hi], w21[2 + hi]);
    c10 = __builtin_amdgcn_wmma_f32_16x16x32_bf16(false, fa, false, f10, (short)0, c10, false, false);
    c11 = __builtin_amdgcn_wmma_f32_16x16x32_bf16(false, fa, false, f11, (short)0, c11, false, false);
    c20 = __builtin_amdgcn_wmma_f32_16x16x32_bf16(false, fa, false, f20, (short)0, c20, false, false);
    c21 = __builtin_amdgcn_wmma_f32_16x16x32_bf16(false, fa, false, f21, (short)0, c21, false, false);
    __syncthreads();
  }
  int nl = lane & 15, hi = lane >> 4;
#pragma unroll
  for (int r = 0; r < 8; ++r) {
    int gm = m0 + wm * 16 + r + hi * 8;
    if (gm >= M) continue;
    int gn0 = n0 + wn * 32 + nl;
    if (gn0 < N) { float x1 = c10[r], x2 = c20[r]; H[(i64)gm * hm + gn0] = f2bf(x1 * sigmoidf_(x1) * x2); }
    int gn1 = gn0 + 16;
    if (gn1 < N) { float x1 = c11[r], x2 = c21[r]; H[(i64)gm * hm + gn1] = f2bf(x1 * sigmoidf_(x1) * x2); }
  }
}

// ---------------- elementwise / reduction kernels ----------------
__global__ __launch_bounds__(256)
void k_layernorm_bf16(const float* __restrict__ x, const float* __restrict__ sc,
                      const float* __restrict__ bs, u16* __restrict__ o, i64 rows, int C)
{
  int lane = threadIdx.x & 31;
  i64 row = (i64)blockIdx.x * 8 + (threadIdx.x >> 5);
  if (row >= rows) return;
  const float* xr = x + row * C;
  float s = 0.f, s2 = 0.f;
  for (int c = lane; c < C; c += 32) { float v = xr[c]; s += v; s2 += v * v; }
  for (int off = 16; off; off >>= 1) { s += __shfl_xor(s, off, 32); s2 += __shfl_xor(s2, off, 32); }
  float mean = s / C;
  float inv = rsqrtf(s2 / C - mean * mean + 1e-5f);
  u16* orow = o + row * C;
  for (int c = lane; c < C; c += 32) orow[c] = f2bf((xr[c] - mean) * inv * sc[c] + bs[c]);
}

// softmax over last dim S (block = S threads), with pair bias bias[(rowlast*S + k)*H + h]
__global__ __launch_bounds__(256)
void k_softmax_bias(const float* __restrict__ lg, const float* __restrict__ bias,
                    u16* __restrict__ o, int H, i64 rows, int S)
{
  i64 r = blockIdx.x;
  int k = threadIdx.x;
  i64 rph = rows / H;
  int h = (int)(r / rph);
  int rl = (int)(r % S);
  float v = lg[r * S + k] + bias[((i64)rl * S + k) * H + h];
  int lane = threadIdx.x & 31, wv = threadIdx.x >> 5;
  __shared__ float red[8];
  float m = v;
  for (int off = 16; off; off >>= 1) m = fmaxf(m, __shfl_xor(m, off, 32));
  if (lane == 0) red[wv] = m;
  __syncthreads();
  float mm = red[0];
#pragma unroll
  for (int i2 = 1; i2 < 8; ++i2) mm = fmaxf(mm, red[i2]);
  float e = __expf(v - mm);
  float ssum = e;
  for (int off = 16; off; off >>= 1) ssum += __shfl_xor(ssum, off, 32);
  __syncthreads();
  if (lane == 0) red[wv] = ssum;
  __syncthreads();
  float tot = 0.f;
#pragma unroll
  for (int i2 = 0; i2 < 8; ++i2) tot += red[i2];
  o[r * S + k] = f2bf(e / tot);
}

__global__ __launch_bounds__(256)
void k_scale_bf16(const float* __restrict__ x, u16* __restrict__ o, i64 n, float s) {
  i64 i = (i64)blockIdx.x * 256 + threadIdx.x;
  if (i < n) o[i] = f2bf(x[i] * s);
}
// fp32 [K,N] weight -> bf16 [N,K] (transposed, k-contiguous)
__global__ __launch_bounds__(256)
void k_cvtw_T(const float* __restrict__ w, u16* __restrict__ o, int K, int N) {
  i64 i = (i64)blockIdx.x * 256 + threadIdx.x;
  if (i >= (i64)K * N) return;
  int n = (int)(i % N);
  int k = (int)(i / N);
  o[(i64)n * K + k] = f2bf(w[i]);
}
// fp32 [row][h*D + d] -> bf16 [row][h*32 + d] zero-padded d in [D,32)
__global__ __launch_bounds__(256)
void k_pad_head(const float* __restrict__ src, u16* __restrict__ o, i64 rows, int H, int D, float sc) {
  i64 i = (i64)blockIdx.x * 256 + threadIdx.x;
  i64 n = rows * H * 32;
  if (i >= n) return;
  int d = (int)(i & 31);
  i64 r = i >> 5;
  int h = (int)(r % H);
  i64 row = r / H;
  o[i] = (d < D) ? f2bf(src[row * (i64)(H * D) + h * D + d] * sc) : (u16)0;
}
__global__ __launch_bounds__(256)
void k_sigmul_bf16(const float* __restrict__ g, const float* __restrict__ v, u16* __restrict__ o, i64 n) {
  i64 i = (i64)blockIdx.x * 256 + threadIdx.x;
  if (i < n) o[i] = f2bf(sigmoidf_(g[i]) * v[i]);
}
__global__ __launch_bounds__(256)
void k_gate_add(float* __restrict__ z, const float* __restrict__ g, const float* __restrict__ y, i64 n) {
  i64 i = (i64)blockIdx.x * 256 + threadIdx.x;
  if (i < n) z[i] += sigmoidf_(g[i]) * y[i];
}
__global__ __launch_bounds__(256)
void k_add(float* __restrict__ a, const float* __restrict__ b, i64 n) {
  i64 i = (i64)blockIdx.x * 256 + threadIdx.x;
  if (i < n) a[i] += b[i];
}
__global__ __launch_bounds__(256)
void k_add_T(float* __restrict__ z, const float* __restrict__ y, int S, int C) {
  i64 i = (i64)blockIdx.x * 256 + threadIdx.x;
  i64 n = (i64)S * S * C;
  if (i >= n) return;
  int c = (int)(i % C);
  i64 x = i / C;
  int row = (int)(x / S), col = (int)(x % S);
  z[i] += y[((i64)col * S + row) * C + c];
}
__global__ __launch_bounds__(256)
void k_transpose_f32(const float* __restrict__ z, float* __restrict__ zt, int S, int C) {
  i64 i = (i64)blockIdx.x * 256 + threadIdx.x;
  i64 n = (i64)S * S * C;
  if (i >= n) return;
  int c = (int)(i % C);
  i64 x = i / C;
  int row = (int)(x / S), col = (int)(x % S);
  zt[i] = z[((i64)col * S + row) * C + c];
}
// [x,y,c] (bf16) -> [c][x*S+y]
__global__ __launch_bounds__(256)
void k_transpose_cik(const u16* __restrict__ in, u16* __restrict__ out, int S, int C) {
  i64 i = (i64)blockIdx.x * 256 + threadIdx.x;
  i64 n = (i64)S * S * C;
  if (i >= n) return;
  int c = (int)(i % C);
  i64 x = i / C;
  out[(i64)c * S * S + x] = in[i];
}
// [x,y,c] (bf16) -> [c][y*S+x]  (plane-transposed channel-major)
__global__ __launch_bounds__(256)
void k_transpose_cik_T(const u16* __restrict__ in, u16* __restrict__ out, int S, int C) {
  i64 i = (i64)blockIdx.x * 256 + threadIdx.x;
  i64 n = (i64)S * S * C;
  if (i >= n) return;
  int c = (int)(i % C);
  i64 x = i / C;
  int a = (int)(x / S), b = (int)(x % S);
  out[(i64)c * S * S + (i64)b * S + a] = in[i];
}
// v fp32 [(i*S+k)*(H*D) + h*D + d]  ->  bf16 [((i*H + h)*D + d)*S + k]   (k-contiguous)
__global__ __launch_bounds__(256)
void k_perm_dT(const float* __restrict__ v, u16* __restrict__ o, int S, int H, int D, int NI) {
  i64 i = (i64)blockIdx.x * 256 + threadIdx.x;
  i64 n = (i64)NI * H * D * S;
  if (i >= n) return;
  int k = (int)(i % S); i64 r = i / S;
  int d = (int)(r % D); r /= D;
  int h = (int)(r % H);
  int ii = (int)(r / H);
  int C = H * D;
  o[i] = f2bf(v[((i64)ii * S + k) * C + h * D + d]);
}

// ---------------- host orchestration ----------------
// Param leaf order = jax pytree (alphabetical) flattening of the params dict.
enum {
  APB_WB = 0, APB_WG, APB_WK, APB_WO, APB_WQ, APB_WV, APB_LNA_B, APB_LNA_S, APB_LNZ_B, APB_LNZ_S,
  PT_W1, PT_W2, PT_W3, PT_LN_B, PT_LN_S,
  ST_W1, ST_W2, ST_W3, ST_LN_B, ST_LN_S,
  TAE_WB, TAE_WG, TAE_WK, TAE_WO, TAE_WQ, TAE_WV, TAE_LN_B, TAE_LN_S,
  TAS_WB, TAS_WG, TAS_WK, TAS_WO, TAS_WQ, TAS_WV, TAS_LN_B, TAS_LN_S,
  TMI_WA, TMI_WAG, TMI_WB, TMI_WBG, TMI_WG, TMI_WO, TMI_LN2_B, TMI_LN2_S, TMI_LN_B, TMI_LN_S,
  TMO_WA, TMO_WAG, TMO_WB, TMO_WBG, TMO_WG, TMO_WO, TMO_LN2_B, TMO_LN2_S, TMO_LN_B, TMO_LN_S,
  N_LEAVES
};
static const i64 LSZ[N_LEAVES] = {
  128 * 16, 384 * 384, 384 * 384, 384 * 384, 384 * 384, 384 * 384, 384, 384, 128, 128,
  128 * 512, 128 * 512, 512 * 128, 128, 128,
  384 * 1536, 384 * 1536, 1536 * 384, 384, 384,
  128 * 4, 128 * 128, 128 * 128, 128 * 128, 128 * 128, 128 * 128, 128, 128,
  128 * 4, 128 * 128, 128 * 128, 128 * 128, 128 * 128, 128 * 128, 128, 128,
  128 * 128, 128 * 128, 128 * 128, 128 * 128, 128 * 128, 128 * 128, 128, 128, 128, 128,
  128 * 128, 128 * 128, 128 * 128, 128 * 128, 128 * 128, 128 * 128, 128, 128, 128, 128
};

extern "C" void kernel_launch(void* const* d_in, const int* in_sizes, int n_in,
                              void* d_out, int out_size, void* d_ws, size_t ws_size,
                              hipStream_t stream) {
  (void)n_in; (void)out_size; (void)ws_size;
  constexpr int S = 256, CZ = 128, CS = 384, HP = 4, DZH = 32, HS = 16, DSH = 24;
  constexpr i64 NR = (i64)S * S;            // 65536 pair rows
  constexpr i64 ZE = NR * CZ;               // z elements
  const float RS32 = 0.17677669529663689f;  // 1/sqrt(32)
  const float RS24 = 0.20412414523193154f;  // 1/sqrt(24)

  // locate inputs (either [s, z, leaves...] or [leaves..., s, z])
  const float* s_in; const float* z_in; int P0;
  if (in_sizes[0] == S * CS) { s_in = (const float*)d_in[0]; z_in = (const float*)d_in[1]; P0 = 2; }
  else { P0 = 0; s_in = (const float*)d_in[N_LEAVES]; z_in = (const float*)d_in[N_LEAVES + 1]; }

  // workspace layout
  char* base = (char*)d_ws;
  size_t off = 0;
  auto alloc = [&](size_t bytes) -> void* {
    void* p = base + off; off += (bytes + 255) & ~(size_t)255; return p;
  };
  float* zcur  = (float*)alloc(ZE * 4);
  float* scur  = (float*)alloc((size_t)S * CS * 4);
  u16*   znb   = (u16*)alloc(ZE * 2);
  float* t0    = (float*)alloc(ZE * 4);
  float* t1    = (float*)alloc(ZE * 4);
  u16*   bufA  = (u16*)alloc(ZE * 2);
  u16*   bufB  = (u16*)alloc(ZE * 2);
  u16*   bufC  = (u16*)alloc(ZE * 2);
  u16*   bufD  = (u16*)alloc(ZE * 2);
  u16*   bufE  = (u16*)alloc(ZE * 2);
  u16*   wb0   = (u16*)alloc((size_t)384 * 1536 * 2);
  u16*   wb1   = (u16*)alloc((size_t)384 * 1536 * 2);
  u16*   wb2   = (u16*)alloc((size_t)1536 * 384 * 2);
  float* biasb = (float*)alloc(NR * HS * 4);
  float* ochnk = (float*)alloc((size_t)32 * S * CZ * 4);
  u16*   snb   = (u16*)alloc((size_t)S * CS * 2);
  u16*   qp    = (u16*)alloc((size_t)S * HS * 32 * 2);   // padded q heads (d=32)
  u16*   kp    = (u16*)alloc((size_t)S * HS * 32 * 2);   // padded k heads (d=32)
  u16*   vbs   = (u16*)alloc((size_t)S * CS * 2);
  u16*   gobs  = (u16*)alloc((size_t)S * CS * 2);
  u16*   hsbuf = (u16*)alloc((size_t)S * 1536 * 2);
  float* gsml  = (float*)alloc((size_t)S * CS * 4);
  float* osml  = (float*)alloc((size_t)S * CS * 4);

  // launch helpers
  auto ew = [&](i64 n) { return dim3((unsigned)((n + 255) / 256)); };
  auto cvt = [&](const float* src, u16* dst, i64 n, float sc) {
    k_scale_bf16<<<ew(n), 256, 0, stream>>>(src, dst, n, sc);
  };
  auto cvtT = [&](const float* w, u16* dst, int K, int N) {
    k_cvtw_T<<<ew((i64)K * N), 256, 0, stream>>>(w, dst, K, N);
  };
  auto ln = [&](const float* x, const float* sc, const float* bs, u16* o, i64 rows, int C) {
    k_layernorm_bf16<<<dim3((unsigned)((rows + 7) / 8)), 256, 0, stream>>>(x, sc, bs, o, rows, C);
  };
  auto gemm = [&](const u16* A, i64 am, i64 ab1, i64 ab2,
                  const u16* B, i64 bn, i64 bb1, i64 bb2,
                  float* C, i64 cm, i64 cn, i64 cb1, i64 cb2,
                  int M, int N, int K, int NB1, int NB2) {
    dim3 g((unsigned)((M + 63) / 64), (unsigned)((N + 63) / 64), (unsigned)(NB1 * NB2));
    k_gemm_bf16<<<g, 256, 0, stream>>>(A, am, ab1, ab2, B, bn, bb1, bb2,
                                       C, cm, cn, cb1, cb2, M, N, K, NB1);
  };
  auto swiglu = [&](const u16* A, i64 am, const u16* W1, const u16* W2,
                    u16* H, i64 hm, int M, int N, int K) {
    dim3 g((unsigned)((M + 63) / 64), (unsigned)((N + 63) / 64), 1);
    k_gemm_swiglu<<<g, 256, 0, stream>>>(A, am, W1, W2, H, hm, M, N, K);
  };

  // init persistent state
  hipMemcpyAsync(zcur, z_in, (size_t)ZE * 4, hipMemcpyDeviceToDevice, stream);
  hipMemcpyAsync(scur, s_in, (size_t)S * CS * 4, hipMemcpyDeviceToDevice, stream);

  for (int blk = 0; blk < 4; ++blk) {
    auto P = [&](int leaf) -> const float* {
      return (const float*)d_in[P0 + leaf] + (i64)blk * LSZ[leaf];
    };

    // ---- triangular multiplicative update (outgoing / incoming) ----
    auto tri_mult = [&](int L, bool outgoing) {
      const float *Wa = P(L + 0), *Wag = P(L + 1), *Wb = P(L + 2), *Wbg = P(L + 3);
      const float *Wg = P(L + 4), *Wo = P(L + 5);
      const float *ln2b = P(L + 6), *ln2s = P(L + 7), *lnb = P(L + 8), *lns = P(L + 9);
      ln(zcur, lns, lnb, znb, NR, CZ);
      // a = sigmoid(zn@Wag) * (zn@Wa)
      cvtT(Wag, wb0, CZ, CZ); cvtT(Wa, wb1, CZ, CZ);
      gemm(znb, CZ, 0, 0, wb0, CZ, 0, 0, t0, CZ, 1, 0, 0, (int)NR, CZ, CZ, 1, 1);
      gemm(znb, CZ, 0, 0, wb1, CZ, 0, 0, t1, CZ, 1, 0, 0, (int)NR, CZ, CZ, 1, 1);
      k_sigmul_bf16<<<ew(ZE), 256, 0, stream>>>(t0, t1, bufA, ZE);
      // b = sigmoid(zn@Wbg) * (zn@Wb)
      cvtT(Wbg, wb0, CZ, CZ); cvtT(Wb, wb1, CZ, CZ);
      gemm(znb, CZ, 0, 0, wb0, CZ, 0, 0, t0, CZ, 1, 0, 0, (int)NR, CZ, CZ, 1, 1);
      gemm(znb, CZ, 0, 0, wb1, CZ, 0, 0, t1, CZ, 1, 0, 0, (int)NR, CZ, CZ, 1, 1);
      k_sigmul_bf16<<<ew(ZE), 256, 0, stream>>>(t0, t1, bufB, ZE);
      // channel-major copies for per-channel batched matmul (plane-transposed for incoming,
      // so both cases end up with k-contiguous A and B operands)
      if (outgoing) {
        k_transpose_cik<<<ew(ZE), 256, 0, stream>>>(bufA, bufC, S, CZ);
        k_transpose_cik<<<ew(ZE), 256, 0, stream>>>(bufB, bufD, S, CZ);
      } else {
        k_transpose_cik_T<<<ew(ZE), 256, 0, stream>>>(bufA, bufC, S, CZ);
        k_transpose_cik_T<<<ew(ZE), 256, 0, stream>>>(bufB, bufD, S, CZ);
      }
      // gate g = zn@Wg (fp32, kept)
      cvtT(Wg, wb0, CZ, CZ);
      gemm(znb, CZ, 0, 0, wb0, CZ, 0, 0, t0, CZ, 1, 0, 0, (int)NR, CZ, CZ, 1, 1);
      // t[i,j,c] = sum_k A_c[i,k] * B_c[j,k]  (both operands k-contiguous)
      gemm(bufC, S, NR, 0, bufD, S, NR, 0, t1, (i64)S * CZ, CZ, 1, 0, S, S, S, CZ, 1);
      // z += sigmoid(g) * (LN2(t) @ Wo)
      ln(t1, ln2s, ln2b, bufC, NR, CZ);
      cvtT(Wo, wb0, CZ, CZ);
      gemm(bufC, CZ, 0, 0, wb0, CZ, 0, 0, t1, CZ, 1, 0, 0, (int)NR, CZ, CZ, 1, 1);
      k_gate_add<<<ew(ZE), 256, 0, stream>>>(zcur, t0, t1, ZE);
    };

    // ---- triangle attention (starting node; ending via transpose) ----
    auto tri_attn = [&](int L, bool transposed) {
      const float *Wb = P(L + 0), *Wg = P(L + 1), *Wk = P(L + 2), *Wo = P(L + 3);
      const float *Wq = P(L + 4), *Wv = P(L + 5), *lnb = P(L + 6), *lns = P(L + 7);
      const float* zsrc = zcur;
      if (transposed) { k_transpose_f32<<<ew(ZE), 256, 0, stream>>>(zcur, t1, S, CZ); zsrc = t1; }
      ln(zsrc, lns, lnb, znb, NR, CZ);
      cvtT(Wq, wb0, CZ, CZ);
      gemm(znb, CZ, 0, 0, wb0, CZ, 0, 0, t0, CZ, 1, 0, 0, (int)NR, CZ, CZ, 1, 1);
      cvt(t0, bufA, ZE, RS32);                                  // q (pre-scaled)
      cvtT(Wk, wb0, CZ, CZ);
      gemm(znb, CZ, 0, 0, wb0, CZ, 0, 0, t0, CZ, 1, 0, 0, (int)NR, CZ, CZ, 1, 1);
      cvt(t0, bufB, ZE, 1.f);                                   // k
      cvtT(Wv, wb0, CZ, CZ);
      gemm(znb, CZ, 0, 0, wb0, CZ, 0, 0, t0, CZ, 1, 0, 0, (int)NR, CZ, CZ, 1, 1);
      k_perm_dT<<<ew(ZE), 256, 0, stream>>>(t0, bufC, S, HP, DZH, S);   // v -> [i][h][d][k]
      cvtT(Wb, wb0, CZ, HP);
      gemm(znb, CZ, 0, 0, wb0, CZ, 0, 0, biasb, HP, 1, 0, 0, (int)NR, HP, CZ, 1, 1);
      cvtT(Wg, wb0, CZ, CZ);
      gemm(znb, CZ, 0, 0, wb0, CZ, 0, 0, t0, CZ, 1, 0, 0, (int)NR, CZ, CZ, 1, 1);  // gate, kept
      const int NI = 32;
      for (int i0 = 0; i0 < S; i0 += NI) {
        i64 co = (i64)i0 * S * CZ;
        // logits[h][il][j][k] = q . k
        gemm(bufA + co, CZ, (i64)S * CZ, DZH,
             bufB + co, CZ, (i64)S * CZ, DZH,
             t1, S, 1, NR, (i64)NI * NR,
             S, S, DZH, NI, HP);
        k_softmax_bias<<<dim3((unsigned)(HP * NI * S)), S, 0, stream>>>(
            t1, biasb, bufD, HP, (i64)HP * NI * S, S);
        // o = attn @ v   (v is k-contiguous)
        gemm(bufD, S, NR, (i64)NI * NR,
             bufC + (i64)i0 * HP * DZH * S, S, (i64)HP * DZH * S, (i64)DZH * S,
             ochnk, CZ, 1, (i64)S * CZ, DZH,
             S, DZH, S, NI, HP);
        k_sigmul_bf16<<<ew((i64)NI * S * CZ), 256, 0, stream>>>(t0 + co, ochnk, bufE + co, (i64)NI * S * CZ);
      }
      cvtT(Wo, wb0, CZ, CZ);
      gemm(bufE, CZ, 0, 0, wb0, CZ, 0, 0, t1, CZ, 1, 0, 0, (int)NR, CZ, CZ, 1, 1);
      if (transposed) k_add_T<<<ew(ZE), 256, 0, stream>>>(zcur, t1, S, CZ);
      else            k_add<<<ew(ZE), 256, 0, stream>>>(zcur, t1, ZE);
    };

    // ---- pair transition (SwiGLU, chunked over rows) ----
    auto transition_z = [&]() {
      ln(zcur, P(PT_LN_S), P(PT_LN_B), znb, NR, CZ);
      const int HD = 4 * CZ;  // 512
      cvtT(P(PT_W1), wb0, CZ, HD);
      cvtT(P(PT_W2), wb1, CZ, HD);
      cvtT(P(PT_W3), wb2, HD, CZ);
      for (i64 m0 = 0; m0 < NR; m0 += 16384) {
        swiglu(znb + m0 * CZ, CZ, wb0, wb1, bufC, HD, 16384, HD, CZ);
        gemm(bufC, HD, 0, 0, wb2, HD, 0, 0, t1 + m0 * CZ, CZ, 1, 0, 0, 16384, CZ, HD, 1, 1);
      }
      k_add<<<ew(ZE), 256, 0, stream>>>(zcur, t1, ZE);
    };

    // ---- attention with pair bias on s ----
    auto apb = [&]() {
      ln(scur, P(APB_LNA_S), P(APB_LNA_B), snb, S, CS);
      ln(zcur, P(APB_LNZ_S), P(APB_LNZ_B), znb, NR, CZ);
      cvtT(P(APB_WQ), wb0, CS, CS);
      gemm(snb, CS, 0, 0, wb0, CS, 0, 0, gsml, CS, 1, 0, 0, S, CS, CS, 1, 1);
      k_pad_head<<<ew((i64)S * HS * 32), 256, 0, stream>>>(gsml, qp, S, HS, DSH, RS24);
      cvtT(P(APB_WK), wb0, CS, CS);
      gemm(snb, CS, 0, 0, wb0, CS, 0, 0, gsml, CS, 1, 0, 0, S, CS, CS, 1, 1);
      k_pad_head<<<ew((i64)S * HS * 32), 256, 0, stream>>>(gsml, kp, S, HS, DSH, 1.f);
      cvtT(P(APB_WV), wb0, CS, CS);
      gemm(snb, CS, 0, 0, wb0, CS, 0, 0, gsml, CS, 1, 0, 0, S, CS, CS, 1, 1);
      k_perm_dT<<<ew((i64)S * CS), 256, 0, stream>>>(gsml, vbs, S, HS, DSH, 1);  // v -> [h][d][j]
      cvtT(P(APB_WB), wb0, CZ, HS);
      gemm(znb, CZ, 0, 0, wb0, CZ, 0, 0, biasb, HS, 1, 0, 0, (int)NR, HS, CZ, 1, 1);
      cvtT(P(APB_WG), wb0, CS, CS);
      gemm(snb, CS, 0, 0, wb0, CS, 0, 0, gsml, CS, 1, 0, 0, S, CS, CS, 1, 1);  // gate, kept
      // logits[h][i][j]  (padded heads: K=32, fast path)
      gemm(qp, (i64)HS * 32, 32, 0, kp, (i64)HS * 32, 32, 0,
           t1, S, 1, NR, 0, S, S, 32, HS, 1);
      k_softmax_bias<<<dim3((unsigned)(HS * S)), S, 0, stream>>>(t1, biasb, bufD, HS, (i64)HS * S, S);
      gemm(bufD, S, NR, 0, vbs, S, (i64)DSH * S, 0, osml, CS, 1, DSH, 0, S, DSH, S, HS, 1);
      k_sigmul_bf16<<<ew((i64)S * CS), 256, 0, stream>>>(gsml, osml, gobs, (i64)S * CS);
      cvtT(P(APB_WO), wb0, CS, CS);
      gemm(gobs, CS, 0, 0, wb0, CS, 0, 0, osml, CS, 1, 0, 0, S, CS, CS, 1, 1);
      k_add<<<ew((i64)S * CS), 256, 0, stream>>>(scur, osml, (i64)S * CS);
    };

    // ---- single-rep transition ----
    auto transition_s = [&]() {
      ln(scur, P(ST_LN_S), P(ST_LN_B), snb, S, CS);
      const int HD = 4 * CS;  // 1536
      cvtT(P(ST_W1), wb0, CS, HD);
      cvtT(P(ST_W2), wb1, CS, HD);
      cvtT(P(ST_W3), wb2, HD, CS);
      swiglu(snb, CS, wb0, wb1, hsbuf, HD, S, HD, CS);
      gemm(hsbuf, HD, 0, 0, wb2, HD, 0, 0, osml, CS, 1, 0, 0, S, CS, HD, 1, 1);
      k_add<<<ew((i64)S * CS), 256, 0, stream>>>(scur, osml, (i64)S * CS);
    };

    tri_mult(TMO_WA, true);
    tri_mult(TMI_WA, false);
    tri_attn(TAS_WB, false);
    tri_attn(TAE_WB, true);
    transition_z();
    apb();
    transition_s();
  }

  float* out = (float*)d_out;
  hipMemcpyAsync(out, scur, (size_t)S * CS * 4, hipMemcpyDeviceToDevice, stream);
  hipMemcpyAsync(out + (i64)S * CS, zcur, (size_t)ZE * 4, hipMemcpyDeviceToDevice, stream);
}
